// RasterizedGaussianGoalMapWithUNetGrid50_1821066134172
// MI455X (gfx1250) — compile-verified, matrix operand order checked by reading, not run on hardware
//
#include <hip/hip_runtime.h>
#include <hip/hip_bf16.h>

// Problem constants (from the reference)
#define BDIM   2048
#define ODIM   48
#define OPAD   64        // K padded to 2 WMMA k-steps of 32
#define GRID_N 112       // H == W == 112 == 7 * 16
#define TILES  7
#define SIGMA  5.0f

typedef _Float16 v16h __attribute__((ext_vector_type(16)));
typedef _Float16 v8h  __attribute__((ext_vector_type(8)));
typedef float    v8f  __attribute__((ext_vector_type(8)));

static __device__ __forceinline__ v16h combine8(v8h lo, v8h hi) {
    v16h r;
#pragma unroll
    for (int i = 0; i < 8; ++i) { r[i] = lo[i]; r[i + 8] = hi[i]; }
    return r;
}

// One workgroup (224 threads = 7 wave32) per batch element.
__global__ __launch_bounds__(224)
void goalmap_wmma_kernel(const float* __restrict__ xL,
                         const float* __restrict__ yL,
                         const float* __restrict__ obj_list,   // [B][O][2] (yo, xo)
                         const int*   __restrict__ obj_num,    // [B]
                         const int*   __restrict__ road_mask,  // [B][H][W]
                         float*       __restrict__ out)        // [B][H][W]
{
    // A[h][o] = valid(o) * exp(-(h-xo)^2 * inv)   (16-byte-aligned rows, o padded to 64)
    // Bt[w][o] = exp(-(w-yo)^2 * inv)             (B stored transposed for contiguous frags)
    __shared__ __align__(16) _Float16 Alds[GRID_N][OPAD];
    __shared__ __align__(16) _Float16 Blds[GRID_N][OPAD];
    __shared__ float gxs[GRID_N];
    __shared__ float gys[GRID_N];
    __shared__ float xos[ODIM];
    __shared__ float yos[ODIM];

    const int b   = blockIdx.x;
    const int tid = threadIdx.x;              // 0..223
    const float inv = 1.0f / (2.0f * SIGMA * SIGMA);

    const size_t base = (size_t)b * (GRID_N * GRID_N);
    const int* mbase = road_mask + base;

    // Warm the road_mask lines for this batch (global_prefetch_b8); 224 threads
    // striding 56 elements (224 B) covers all 50 KB of the mask tile.
    __builtin_prefetch(mbase + tid * 56, 0, 1);

    // ---- Stage per-batch scalars / vectors ---------------------------------
    const float xLb = xL[b];
    const float yLb = yL[b];
    const int   nObj = obj_num[b];

    if (tid < ODIM) {
        yos[tid] = obj_list[(size_t)b * (ODIM * 2) + tid * 2 + 0];
        xos[tid] = obj_list[(size_t)b * (ODIM * 2) + tid * 2 + 1];
    }
    if (tid < GRID_N) {
        float d = (float)tid - xLb;
        gxs[tid] = __expf(-d * d * inv);
    } else if (tid < 2 * GRID_N) {
        int w = tid - GRID_N;
        float d = (float)w - yLb;
        gys[w] = __expf(-d * d * inv);
    }
    __syncthreads();

    // ---- Build A (H x Opad) and Bt (W x Opad) operands in LDS --------------
#pragma unroll 4
    for (int e = tid; e < GRID_N * OPAD; e += 224) {
        const int h = e >> 6;          // 0..111  (also used as w for Bt)
        const int o = e & 63;          // 0..63
        float av = 0.0f, bv = 0.0f;
        if (o < ODIM) {
            const float vmask = (o < nObj) ? 1.0f : 0.0f;
            float dx = (float)h - xos[o];
            av = vmask * __expf(-dx * dx * inv);
            float dy = (float)h - yos[o];
            bv = __expf(-dy * dy * inv);
        }
        Alds[h][o] = (_Float16)av;
        Blds[h][o] = (_Float16)bv;
    }
    __syncthreads();

    // ---- GEMM: obj = A(112x48) x B(48x112), one tile-row per wave ----------
    const int wave    = tid >> 5;      // 0..6 -> tile-row
    const int lane    = tid & 31;
    const int half16  = lane >> 4;     // 0: lanes 0-15, 1: lanes 16-31
    const int rowSel  = lane & 15;     // M (A) / N (B,D) index within tile
    const int kBase   = half16 * 8;    // per-ISA 16-bit fragment K base

    const int m0 = wave * 16;
    const int ar = m0 + rowSel;

    // A fragments for k-steps K=[0,32) and K=[32,64): each = two ds_load_b128
    const v16h a0 = combine8(*(const v8h*)&Alds[ar][kBase +  0],
                             *(const v8h*)&Alds[ar][kBase + 16]);
    const v16h a1 = combine8(*(const v8h*)&Alds[ar][kBase + 32],
                             *(const v8h*)&Alds[ar][kBase + 48]);

#pragma unroll
    for (int tn = 0; tn < TILES; ++tn) {
        const int n  = tn * 16 + rowSel;

        const v16h b0 = combine8(*(const v8h*)&Blds[n][kBase +  0],
                                 *(const v8h*)&Blds[n][kBase + 16]);
        const v16h b1 = combine8(*(const v8h*)&Blds[n][kBase + 32],
                                 *(const v8h*)&Blds[n][kBase + 48]);

        v8f acc = {};
        acc = __builtin_amdgcn_wmma_f32_16x16x32_f16(
                  false, a0, false, b0, (short)0, acc, false, false);
        acc = __builtin_amdgcn_wmma_f32_16x16x32_f16(
                  false, a1, false, b1, (short)0, acc, false, false);

        // Fused epilogue: channel = 0.5 + 0.5*gx*gy - 0.5*obj, masked.
        const float gyv = 0.5f * gys[n];
#pragma unroll
        for (int i = 0; i < 8; ++i) {
            const int m = m0 + i + half16 * 8;           // C/D layout row
            const size_t idx = base + (size_t)m * GRID_N + n;
            float val = 0.5f + gxs[m] * gyv - 0.5f * acc[i];
            val = (mbase[(size_t)m * GRID_N + n] != 0) ? val : 0.0f;
            out[idx] = val;
        }
    }
}

extern "C" void kernel_launch(void* const* d_in, const int* in_sizes, int n_in,
                              void* d_out, int out_size, void* d_ws, size_t ws_size,
                              hipStream_t stream) {
    const float* xL       = (const float*)d_in[0];
    const float* yL       = (const float*)d_in[1];
    const float* obj_list = (const float*)d_in[2];
    const int*   obj_num  = (const int*)d_in[3];
    const int*   road_msk = (const int*)d_in[4];
    float* out = (float*)d_out;

    const int B = in_sizes[0];   // 2048
    dim3 grid(B), block(224);
    goalmap_wmma_kernel<<<grid, block, 0, stream>>>(xL, yL, obj_list, obj_num,
                                                    road_msk, out);
}
